// GraphSAGE_23699629539648
// MI455X (gfx1250) — compile-verified, matrix operand order checked by reading, not run on hardware
//
#include <hip/hip_runtime.h>
#include <hip/hip_bf16.h>
#include <cstdint>
#include <cstddef>

// ---------------------------------------------------------------------------
// GraphSAGE forward for MI455X (gfx1250):
//   wave32 WMMA bf16 GEMMs + async global->LDS staging of weight tiles.
// ---------------------------------------------------------------------------

typedef __attribute__((ext_vector_type(16))) __bf16 v16bf;
typedef __attribute__((ext_vector_type(8)))  __bf16 v8bf;
typedef __attribute__((ext_vector_type(8)))  float  v8f;

#define BN_EPS 1e-5f

// Native RNE conversion -> lets clang use V_CVT_PK_BF16_F32 on gfx1250.
__device__ __forceinline__ __bf16 f2bf(float f) { return (__bf16)f; }

// 16-byte async global->LDS copy (ASYNCcnt path). The low 32 bits of a
// generic shared pointer are the LDS byte offset (ISA 10.2 aperture rule).
__device__ __forceinline__ void async_copy_b128(const void* gptr, void* lptr) {
  uint32_t lds = (uint32_t)(uintptr_t)lptr;
  asm volatile("global_load_async_to_lds_b128 %0, %1, off"
               :: "v"(lds), "v"(gptr) : "memory");
}
__device__ __forceinline__ void wait_async() {
  asm volatile("s_wait_asynccnt 0" ::: "memory");
}

// ---------------------------------------------------------------------------
// Pack fp32 weights [K x HA] (+ optional [K x HB]) into bf16 TRANSPOSED
// [(HA+HB) x Kpad], zero-padded in K, so B tiles are contiguous 64B runs per
// output column and can be staged with async b128 copies (no transpose in LDS).
// ---------------------------------------------------------------------------
__global__ void pack_wt(__bf16* __restrict__ dst,
                        const float* __restrict__ A, const float* __restrict__ B,
                        int K, int Kpad, int HA, int HB) {
  int ncols = HA + HB;
  long idx = (long)blockIdx.x * blockDim.x + threadIdx.x;
  if (idx >= (long)ncols * Kpad) return;
  int c = (int)(idx / Kpad), k = (int)(idx % Kpad);
  float v = 0.f;
  if (k < K) v = (c < HA) ? A[(size_t)k * HA + c] : B[(size_t)k * HB + (c - HA)];
  dst[idx] = f2bf(v);
}

// ---------------------------------------------------------------------------
// One K-step of WMMA: build A/B fragments from LDS per the 16-bit 16x32
// layout (element i -> k = kb + i + (i&8)) and accumulate.
// ---------------------------------------------------------------------------
template <int NCOLS>
__device__ __forceinline__ void wmma_step(const __bf16* __restrict__ sA,
                                          const __bf16* __restrict__ sBt,
                                          int wM, int wN, int mlow, int kb,
                                          v8f (&acc)[2][NCOLS / 64]) {
  constexpr int WCOLS = NCOLS / 4;
  constexpr int NACC  = NCOLS / 64;
  v16bf a[2], b[NACC];
#pragma unroll
  for (int mi = 0; mi < 2; ++mi) {
    const __bf16* p = &sA[(wM * 32 + mi * 16 + mlow) * 32 + kb];
    v8bf lo = *(const v8bf*)p;
    v8bf hi = *(const v8bf*)(p + 16);
    a[mi] = __builtin_shufflevector(lo, hi, 0,1,2,3,4,5,6,7,8,9,10,11,12,13,14,15);
  }
#pragma unroll
  for (int ni = 0; ni < NACC; ++ni) {
    const __bf16* p = &sBt[(wN * WCOLS + ni * 16 + mlow) * 32 + kb];
    v8bf lo = *(const v8bf*)p;
    v8bf hi = *(const v8bf*)(p + 16);
    b[ni] = __builtin_shufflevector(lo, hi, 0,1,2,3,4,5,6,7,8,9,10,11,12,13,14,15);
  }
#pragma unroll
  for (int mi = 0; mi < 2; ++mi)
#pragma unroll
    for (int ni = 0; ni < NACC; ++ni)
      acc[mi][ni] = __builtin_amdgcn_wmma_f32_16x16x32_bf16(
          false, a[mi], false, b[ni], (short)0, acc[mi][ni], false, false);
}

// ---------------------------------------------------------------------------
// GEMM: C[N x NCOLS] = X[N x K] (fp32, ld=ldx) @ W (bf16, transposed
// [NCOLS x Kpad]).  256 threads = 8 waves, block tile 64 x NCOLS,
// wave grid 2(M) x 4(N), per-wave 2 x (NCOLS/64) 16x16 f32 accumulators.
// A: fp32->bf16 convert (V_CVT_PK_BF16_F32), packed v8bf LDS stores,
//    unguarded main loop (row index clamped; stores guarded instead).
// B: async global->LDS b128 copies drained with s_wait_asynccnt.
// ---------------------------------------------------------------------------
template <int NCOLS>
__global__ __launch_bounds__(256) void gemm_bf16(
    const float* __restrict__ X, int nrows, int K, int Kpad, int ldx,
    const __bf16* __restrict__ Wt, float* __restrict__ C) {
  constexpr int TILE_M = 64;
  constexpr int WCOLS  = NCOLS / 4;
  constexpr int NACC   = NCOLS / 64;

  __shared__ __align__(16) __bf16 sA[TILE_M * 32];
  __shared__ __align__(16) __bf16 sBt[NCOLS * 32];

  const int tix  = threadIdx.x;
  const int lane = tix & 31;
  const int wid  = tix >> 5;
  const int wM   = wid >> 2;
  const int wN   = wid & 3;
  const int rowBase = blockIdx.x * TILE_M;
  const int mlow = lane & 15;
  const int kb   = (lane >> 4) * 8;

  // A staging (main path): 4 threads per row, 8 consecutive k each; clamp row.
  const int ar = tix >> 2;
  const int ak = (tix & 3) * 8;
  int arg = rowBase + ar;
  if (arg >= nrows) arg = nrows - 1;
  const float* Xrow = X + (size_t)arg * ldx;

  v8f zero = {0.f, 0.f, 0.f, 0.f, 0.f, 0.f, 0.f, 0.f};
  v8f acc[2][NACC];
#pragma unroll
  for (int mi = 0; mi < 2; ++mi)
#pragma unroll
    for (int ni = 0; ni < NACC; ++ni) acc[mi][ni] = zero;

  const int Kmain = K & ~31;
  int kk = 0;
  for (; kk < Kmain; kk += 32) {
    __syncthreads();
    {
      const float4 f0 = *(const float4*)(Xrow + kk + ak);
      const float4 f1 = *(const float4*)(Xrow + kk + ak + 4);
      v8bf pk;
      pk[0] = f2bf(f0.x); pk[1] = f2bf(f0.y); pk[2] = f2bf(f0.z); pk[3] = f2bf(f0.w);
      pk[4] = f2bf(f1.x); pk[5] = f2bf(f1.y); pk[6] = f2bf(f1.z); pk[7] = f2bf(f1.w);
      *(v8bf*)&sA[ar * 32 + ak] = pk;
      if (kk + 32 < K)  // speculative prefetch of next K tile
        __builtin_prefetch(Xrow + kk + 32 + ak, 0, 3);
    }
    // B tile: NCOLS x 32 bf16 = NCOLS*4 chunks of 16B, async to LDS.
#pragma unroll
    for (int j = 0; j < (NCOLS * 4) / 256; ++j) {
      int idx = tix + j * 256;
      int n = idx >> 2, c = (idx & 3) * 8;
      async_copy_b128(Wt + (size_t)n * Kpad + kk + c, &sBt[n * 32 + c]);
    }
    wait_async();
    __syncthreads();
    wmma_step<NCOLS>(sA, sBt, wM, wN, mlow, kb, acc);
  }

  if (kk < Kpad) {  // guarded K tail (layer 0: 700 % 32 == 28)
    __syncthreads();
#pragma unroll
    for (int j = 0; j < (TILE_M * 32) / 256; ++j) {
      int idx = tix + j * 256;
      int r = idx >> 5, k = idx & 31;
      int rg = rowBase + r;
      float v = 0.f;
      if (rg < nrows && (kk + k) < K) v = X[(size_t)rg * ldx + kk + k];
      sA[r * 32 + k] = f2bf(v);
    }
#pragma unroll
    for (int j = 0; j < (NCOLS * 4) / 256; ++j) {
      int idx = tix + j * 256;
      int n = idx >> 2, c = (idx & 3) * 8;
      async_copy_b128(Wt + (size_t)n * Kpad + kk + c, &sBt[n * 32 + c]);
    }
    wait_async();
    __syncthreads();
    wmma_step<NCOLS>(sA, sBt, wM, wN, mlow, kb, acc);
  }

  // Store: C/D layout -> VGPR r holds M = r + (lane>=16)*8, N = lane&15.
  const int mhigh = (lane >> 4) * 8;
#pragma unroll
  for (int mi = 0; mi < 2; ++mi) {
#pragma unroll
    for (int r = 0; r < 8; ++r) {
      int row = rowBase + wM * 32 + mi * 16 + mhigh + r;
      if (row < nrows) {
#pragma unroll
        for (int ni = 0; ni < NACC; ++ni) {
          int col = wN * WCOLS + ni * 16 + (lane & 15);
          C[(size_t)row * NCOLS + col] = acc[mi][ni][r];
        }
      }
    }
  }
}

// ---------------------------------------------------------------------------
// Degree / reciprocal
// ---------------------------------------------------------------------------
__global__ void deg_kernel(const int* __restrict__ edges, float* __restrict__ deg, int E) {
  int e = blockIdx.x * blockDim.x + threadIdx.x;
  if (e < E) atomicAdd(&deg[edges[2 * e + 1]], 1.f);
}
__global__ void recip_kernel(float* __restrict__ d, int n) {
  int i = blockIdx.x * blockDim.x + threadIdx.x;
  if (i < n) d[i] = 1.f / fmaxf(d[i], 1.f);
}

// ---------------------------------------------------------------------------
// agg[dst, c] += g[src, 128 + c]  (hn half of the fused GEMM output, stride 256)
// ---------------------------------------------------------------------------
__global__ void scatter_add(const int* __restrict__ edges, const float* __restrict__ g,
                            float* __restrict__ agg, int E) {
  long tid = (long)blockIdx.x * blockDim.x + threadIdx.x;
  int e = (int)(tid >> 5);
  if (e >= E) return;
  int c4 = ((int)tid & 31) * 4;
  int s = edges[2 * e + 0];
  int d = edges[2 * e + 1];
  const float4 v = *(const float4*)(g + (size_t)s * 256 + 128 + c4);
  float* a = agg + (size_t)d * 128 + c4;
  atomicAdd(a + 0, v.x); atomicAdd(a + 1, v.y);
  atomicAdd(a + 2, v.z); atomicAdd(a + 3, v.w);
}

// ---------------------------------------------------------------------------
// t = g_self + agg*recip + bias ; accumulate per-channel sum / sumsq for BN.
// ---------------------------------------------------------------------------
__global__ void combine_stats(const float* __restrict__ g, int gs,
                              const float* __restrict__ agg, const float* __restrict__ recip,
                              const float* __restrict__ bias,
                              float* __restrict__ t, int ts,
                              float* __restrict__ sums, int n, int CH) {
  __shared__ float ssum[128];
  __shared__ float ssq[128];
  int tix = threadIdx.x;
  if (tix < CH) { ssum[tix] = 0.f; ssq[tix] = 0.f; }
  __syncthreads();
  long idx = (long)blockIdx.x * blockDim.x + tix;
  if (idx < (long)n * CH) {
    int i = (int)(idx / CH), c = (int)(idx % CH);
    float v = g[(size_t)i * gs + c] + bias[c];
    if (agg) v += agg[(size_t)i * CH + c] * recip[i];
    t[(size_t)i * ts + c] = v;
    atomicAdd(&ssum[c], v);
    atomicAdd(&ssq[c], v * v);
  }
  __syncthreads();
  if (tix < CH) {
    atomicAdd(&sums[tix], ssum[tix]);
    atomicAdd(&sums[CH + tix], ssq[tix]);
  }
}

__global__ void bn_finalize(const float* __restrict__ sums, int n,
                            const float* __restrict__ gamma, const float* __restrict__ beta,
                            float* __restrict__ scl, float* __restrict__ shf, int CH) {
  int c = threadIdx.x;
  if (c >= CH) return;
  float inv_n = 1.f / (float)n;
  float mean = sums[c] * inv_n;
  float var = fmaxf(sums[CH + c] * inv_n - mean * mean, 0.f);
  float s = gamma[c] * rsqrtf(var + BN_EPS);
  scl[c] = s;
  shf[c] = beta[c] - mean * s;
}

// out = relu(t*scale + shift [+ resid])
__global__ void apply_bn(const float* __restrict__ t, int ts,
                         const float* __restrict__ scl, const float* __restrict__ shf,
                         const float* __restrict__ resid,
                         float* __restrict__ out, int n, int CH) {
  long idx = (long)blockIdx.x * blockDim.x + threadIdx.x;
  if (idx >= (long)n * CH) return;
  int i = (int)(idx / CH), c = (int)(idx % CH);
  float v = t[(size_t)i * ts + c] * scl[c] + shf[c];
  if (resid) v += resid[(size_t)i * CH + c];
  out[(size_t)i * CH + c] = fmaxf(v, 0.f);
}

// Fused head: relu(bn(z)) @ W2 + b2, one node per thread (z row is 64 floats).
__global__ void head_out(const float* __restrict__ z, const float* __restrict__ scl,
                         const float* __restrict__ shf, const float* __restrict__ W2,
                         const float* __restrict__ b2, float* __restrict__ out, int n) {
  int i = blockIdx.x * blockDim.x + threadIdx.x;
  if (i >= n) return;
  float acc = b2[0];
#pragma unroll
  for (int c = 0; c < 64; ++c) {
    float v = z[(size_t)i * 64 + c] * scl[c] + shf[c];
    acc += fmaxf(v, 0.f) * W2[c];
  }
  out[i] = acc;
}

// ---------------------------------------------------------------------------
static inline int cdiv(long a, long b) { return (int)((a + b - 1) / b); }

extern "C" void kernel_launch(void* const* d_in, const int* in_sizes, int n_in,
                              void* d_out, int out_size, void* d_ws, size_t ws_size,
                              hipStream_t stream) {
  const float* X       = (const float*)d_in[0];
  const int*   graph   = (const int*)d_in[1];
  const float* Wself0  = (const float*)d_in[2];
  const float* Wneigh0 = (const float*)d_in[3];
  const float* b0      = (const float*)d_in[4];
  const float* Wself   = (const float*)d_in[5];
  const float* Wneigh  = (const float*)d_in[6];
  const float* bvec    = (const float*)d_in[7];
  const float* gam     = (const float*)d_in[8];
  const float* bet     = (const float*)d_in[9];
  const float* W1      = (const float*)d_in[10];
  const float* b1      = (const float*)d_in[11];
  const float* bng1    = (const float*)d_in[12];
  const float* bnb1    = (const float*)d_in[13];
  const float* W2      = (const float*)d_in[14];
  const float* b2      = (const float*)d_in[15];
  float* out = (float*)d_out;

  const int FIN = 700, H = 128;
  const int N = in_sizes[0] / FIN;
  const int E = in_sizes[1] / 2;
  const int K0PAD = 704;  // 700 rounded to 32

  uintptr_t p = (uintptr_t)d_ws;
  auto take = [&](size_t bytes) -> void* {
    uintptr_t r = (p + 255) & ~(uintptr_t)255;
    p = r + bytes;
    return (void*)r;
  };
  float*  recip = (float*)take((size_t)N * 4);
  float*  g     = (float*)take((size_t)N * 256 * 4);
  float*  agg   = (float*)take((size_t)N * 128 * 4);
  float*  hA    = (float*)take((size_t)N * 128 * 4);
  float*  hB    = (float*)take((size_t)N * 128 * 4);
  float*  sums  = (float*)take(256 * 4);
  float*  scl   = (float*)take(128 * 4);
  float*  shf   = (float*)take(128 * 4);
  __bf16* Wp    = (__bf16*)take((size_t)256 * K0PAD * 2);  // transposed weights
  float*  z     = agg;  // head activations alias agg (dead by then)

  hipMemsetAsync(recip, 0, (size_t)N * 4, stream);
  deg_kernel<<<cdiv(E, 256), 256, 0, stream>>>(graph, recip, E);
  recip_kernel<<<cdiv(N, 256), 256, 0, stream>>>(recip, N);

  const int gemm_grid = cdiv(N, 64);
  const int nh_grid   = cdiv((long)N * H, 256);
  const int sc_grid   = cdiv((long)E * 32, 256);

  // ---- Layer 0: X @ [Wself0 | Wneigh0] ----
  pack_wt<<<cdiv((long)256 * K0PAD, 256), 256, 0, stream>>>(Wp, Wself0, Wneigh0, FIN, K0PAD, H, H);
  gemm_bf16<256><<<gemm_grid, 256, 0, stream>>>(X, N, FIN, K0PAD, FIN, Wp, g);
  hipMemsetAsync(agg, 0, (size_t)N * 128 * 4, stream);
  scatter_add<<<sc_grid, 256, 0, stream>>>(graph, g, agg, E);
  hipMemsetAsync(sums, 0, 256 * 4, stream);
  combine_stats<<<nh_grid, 256, 0, stream>>>(g, 256, agg, recip, b0, g, 256, sums, N, H);
  bn_finalize<<<1, 128, 0, stream>>>(sums, N, gam + 0 * H, bet + 0 * H, scl, shf, H);
  apply_bn<<<nh_grid, 256, 0, stream>>>(g, 256, scl, shf, nullptr, hA, N, H);

  // ---- Layers 1..3 (residual) ----
  float* hCur = hA;
  float* hNext = hB;
  for (int l = 0; l < 3; ++l) {
    pack_wt<<<cdiv((long)256 * H, 256), 256, 0, stream>>>(
        Wp, Wself + (size_t)l * H * H, Wneigh + (size_t)l * H * H, H, H, H, H);
    gemm_bf16<256><<<gemm_grid, 256, 0, stream>>>(hCur, N, H, H, H, Wp, g);
    hipMemsetAsync(agg, 0, (size_t)N * 128 * 4, stream);
    scatter_add<<<sc_grid, 256, 0, stream>>>(graph, g, agg, E);
    hipMemsetAsync(sums, 0, 256 * 4, stream);
    combine_stats<<<nh_grid, 256, 0, stream>>>(g, 256, agg, recip, bvec + (size_t)l * H,
                                               g, 256, sums, N, H);
    bn_finalize<<<1, 128, 0, stream>>>(sums, N, gam + (size_t)(l + 1) * H,
                                       bet + (size_t)(l + 1) * H, scl, shf, H);
    apply_bn<<<nh_grid, 256, 0, stream>>>(g, 256, scl, shf, hCur, hNext, N, H);
    float* tmp = hCur; hCur = hNext; hNext = tmp;
  }

  // ---- Head: z = h @ W1 (128x64), BN, relu, @ W2 + b2 ----
  pack_wt<<<cdiv((long)64 * H, 256), 256, 0, stream>>>(Wp, W1, nullptr, H, H, 64, 0);
  gemm_bf16<64><<<gemm_grid, 256, 0, stream>>>(hCur, N, H, H, H, Wp, z);
  hipMemsetAsync(sums, 0, 128 * 4, stream);
  combine_stats<<<cdiv((long)N * 64, 256), 256, 0, stream>>>(z, 64, nullptr, nullptr, b1,
                                                             z, 64, sums, N, 64);
  bn_finalize<<<1, 64, 0, stream>>>(sums, N, bng1, bnb1, scl, shf, 64);
  head_out<<<cdiv(N, 256), 256, 0, stream>>>(z, scl, shf, W2, b2, out, N);
}
